// CrissCrossAttention_90314572300386
// MI455X (gfx1250) — matrix operand order repaired and checked by reading, not compiled, not run on hardware
//
#include <hip/hip_runtime.h>

typedef _Float16 v16h __attribute__((ext_vector_type(16)));
typedef _Float16 v8h  __attribute__((ext_vector_type(8)));
typedef float    v8f  __attribute__((ext_vector_type(8)));

#define NEG_INF (-1e30f)
#define B_   4
#define C_   256
#define H_   96
#define W_   96
#define CQ_  32
#define HP_  102     // padded spatial (halo 3 each side)
#define NT_  27      // 3 dilations x 9 taps

// Flip to 0 if the async-to-LDS inline asm is rejected by the assembler.
#define USE_ASYNC_STAGE 1

// ---------------------------------------------------------------------------
// WMMA helpers (gfx1250, wave32). D = A(16x32) x B(32x16) + C, f16 in / f32 acc
// ---------------------------------------------------------------------------
__device__ __forceinline__ v8f wmma32(v16h a, v16h b, v8f c) {
    return __builtin_amdgcn_wmma_f32_16x16x32_f16(
        /*neg_a=*/false, a, /*neg_b=*/false, b,
        /*c_mod=*/(short)0, c, /*reuse_a=*/false, /*reuse_b=*/false);
}

// A fragment: 16x32 f16 tile, row-major rows of stride ld (halfs, mult of 8).
// lane m (0..15)/half hf: elems 0..7 = K hf*8.., elems 8..15 = K 16+hf*8..
__device__ __forceinline__ v16h frag_a(const _Float16* __restrict__ p, int ld) {
    const int lane = threadIdx.x & 31;
    const int m = lane & 15, hf = lane >> 4;
    const _Float16* b0 = p + m * ld + hf * 8;
    v8h lo = *reinterpret_cast<const v8h*>(b0);
    v8h hi = *reinterpret_cast<const v8h*>(b0 + 16);
    v16h f;
#pragma unroll
    for (int i = 0; i < 8; ++i) { f[i] = lo[i]; f[8 + i] = hi[i]; }
    return f;
}

// B fragment: 32x16 tile stored COLUMN-major ([n][k], column stride ld halfs).
// lane n (0..15)/half hf: elems 0..7 = K hf*8.., elems 8..15 = K 16+hf*8..
__device__ __forceinline__ v16h frag_b(const _Float16* __restrict__ p, int ld) {
    const int lane = threadIdx.x & 31;
    const int n = lane & 15, hf = lane >> 4;
    const _Float16* b0 = p + n * ld + hf * 8;
    v8h lo = *reinterpret_cast<const v8h*>(b0);
    v8h hi = *reinterpret_cast<const v8h*>(b0 + 16);
    v16h f;
#pragma unroll
    for (int i = 0; i < 8; ++i) { f[i] = lo[i]; f[8 + i] = hi[i]; }
    return f;
}

// ---------------------------------------------------------------------------
// Prep kernels: f32 -> f16 repacks
// ---------------------------------------------------------------------------
__global__ void k_pad(const float* __restrict__ x, _Float16* __restrict__ xp) {
    int idx = blockIdx.x * 256 + threadIdx.x;        // over B*HP*HP*C
    int c = idx & 255;
    int t = idx >> 8;
    int wp = t % HP_; t /= HP_;
    int hp = t % HP_; int b = t / HP_;
    if (b >= B_) return;
    float v = 0.f;
    int h = hp - 3, w = wp - 3;
    if (h >= 0 && h < H_ && w >= 0 && w < W_)
        v = x[((b * C_ + c) * H_ + h) * W_ + w];
    xp[idx] = (_Float16)v;
}

__global__ void k_packw(const float* __restrict__ wms, _Float16* __restrict__ wpk) {
    int idx = blockIdx.x * 256 + threadIdx.x;        // over 27*256*256, [t][n][k]
    int k = idx & 255;
    int t2 = idx >> 8;
    int n = t2 & 255;
    int t = t2 >> 8;
    if (t >= NT_) return;
    int d = t / 9, r = t % 9, ky = r / 3, kx = r % 3;
    wpk[idx] = (_Float16)wms[(((d * C_ + n) * C_ + k) * 3 + ky) * 3 + kx];
}

__global__ void k_packqkv(const float* __restrict__ wq, const float* __restrict__ bq,
                          const float* __restrict__ wk, const float* __restrict__ bk,
                          const float* __restrict__ wv, const float* __restrict__ bv,
                          const float* __restrict__ bms0,
                          _Float16* __restrict__ wqkv, float* __restrict__ bqkv,
                          float* __restrict__ bms) {
    int idx = blockIdx.x * 256 + threadIdx.x;        // over 320*256, [j][k]
    if (idx >= 320 * 256) return;
    int k = idx & 255, j = idx >> 8;
    float v = (j < 32) ? wq[j * 256 + k]
            : (j < 64) ? wk[(j - 32) * 256 + k]
                       : wv[(j - 64) * 256 + k];
    wqkv[idx] = (_Float16)v;
    if (k == 0)
        bqkv[j] = (j < 32) ? bq[j] : (j < 64) ? bk[j - 32] : bv[j - 64];
    if (idx < 256)
        bms[idx] = bms0[idx] + bms0[256 + idx] + bms0[512 + idx];
}

// ---------------------------------------------------------------------------
// Multi-scale dilated conv as implicit GEMM with 2x2 register blocking.
// Block = 256 thr (8 waves), covers 32 pixels x 256 out-channels.
// Wave w owns N tiles {w*32, w*32+16} x M tiles {0,16} -> 4 accumulators,
// 4 WMMAs per fragment-load group (32 B of fragments per WMMA).
// A tile (32 px x 256 ch = 16 KB contiguous) staged per tap via async-to-LDS.
// ---------------------------------------------------------------------------
__global__ __launch_bounds__(256)
void k_conv(const _Float16* __restrict__ xp, const _Float16* __restrict__ wpk,
            const float* __restrict__ bms, _Float16* __restrict__ ms) {
    __shared__ __align__(16) _Float16 As[32 * C_];   // 16 KB A-tile
    int tile = blockIdx.x;                           // 0..1151
    int b = tile / (H_ * 3);
    int r = tile % (H_ * 3);
    int h = r / 3, w0 = (r % 3) * 32;
    int tid = threadIdx.x;
    int n0 = (tid >> 5) * 32;
    v8f acc00 = {}, acc01 = {}, acc10 = {}, acc11 = {};
#if USE_ASYNC_STAGE
    typedef __attribute__((address_space(3))) _Float16 lds_f16;
    unsigned lds_base = (unsigned)(uintptr_t)(lds_f16*)&As[0];
#endif
    for (int t = 0; t < NT_; ++t) {
        int dil = t / 9 + 1, rr = t % 9;
        int dy = (rr / 3 - 1) * dil, dx = (rr % 3 - 1) * dil;
        const _Float16* src =
            xp + (((b * HP_ + (h + 3 + dy)) * HP_ + (w0 + 3 + dx)) * C_);
        // stage 32 pixels x 256 channels: 16 B per thread per chunk, 4 chunks
#if USE_ASYNC_STAGE
        {
            const _Float16* g = src + tid * 32;
            unsigned l = lds_base + tid * 64;
#pragma unroll
            for (int j = 0; j < 4; ++j)
                asm volatile("global_load_async_to_lds_b128 %0, %1, off"
                             :: "v"(l + j * 16),
                                "v"((unsigned long long)(uintptr_t)(g + j * 8))
                             : "memory");
        }
        asm volatile("s_wait_asynccnt 0x0" ::: "memory");
#else
#pragma unroll
        for (int j = 0; j < 4; ++j)
            *reinterpret_cast<v8h*>(&As[tid * 32 + j * 8]) =
                *reinterpret_cast<const v8h*>(src + tid * 32 + j * 8);
#endif
        __syncthreads();
        const _Float16* wt = wpk + (t * C_ + n0) * C_;
        if (t + 1 < NT_)     // warm WGP$ for next tap's weights
            __builtin_prefetch(wpk + ((t + 1) * C_ + n0) * C_ + (tid & 31) * 64, 0, 0);
#pragma unroll
        for (int kc = 0; kc < 8; ++kc) {
            v16h a0 = frag_a(&As[kc * 32], C_);
            v16h a1 = frag_a(&As[16 * C_ + kc * 32], C_);
            v16h b0 = frag_b(wt + kc * 32, C_);
            v16h b1 = frag_b(wt + 16 * C_ + kc * 32, C_);
            acc00 = wmma32(a0, b0, acc00);
            acc01 = wmma32(a0, b1, acc01);
            acc10 = wmma32(a1, b0, acc10);
            acc11 = wmma32(a1, b1, acc11);
        }
        __syncthreads();
    }
    int lane = tid & 31, col = lane & 15, hf = lane >> 4;
    int base_pix = (b * H_ + h) * W_ + w0;
    int na = n0 + col, nb = n0 + 16 + col;
    float ba = bms[na], bb = bms[nb];
#pragma unroll
    for (int i = 0; i < 8; ++i) {
        int wrow = i + hf * 8;
        ms[(base_pix + wrow) * C_ + na]      = (_Float16)(acc00[i] + ba);
        ms[(base_pix + wrow) * C_ + nb]      = (_Float16)(acc01[i] + bb);
        ms[(base_pix + 16 + wrow) * C_ + na] = (_Float16)(acc10[i] + ba);
        ms[(base_pix + 16 + wrow) * C_ + nb] = (_Float16)(acc11[i] + bb);
    }
}

// ---------------------------------------------------------------------------
// Fused q/k/v 1x1-conv GEMM. grid = (2304 pixel tiles, 20 out tiles), 1 wave.
// j<32 -> q, j<64 -> k, else v written to BOTH transposed layouts.
// ---------------------------------------------------------------------------
__global__ __launch_bounds__(32)
void k_qkv(const _Float16* __restrict__ ms, const _Float16* __restrict__ wqkv,
           const float* __restrict__ bqkv, _Float16* __restrict__ q,
           _Float16* __restrict__ kk, _Float16* __restrict__ vW,
           _Float16* __restrict__ vH) {
    int tile = blockIdx.x;
    int b = tile / (H_ * 6);
    int r = tile % (H_ * 6);
    int h = r / 6, w0 = (r % 6) * 16;
    int j0 = blockIdx.y * 16;
    const _Float16* arow = ms + ((b * H_ + h) * W_ + w0) * C_;
    const _Float16* wt = wqkv + j0 * C_;
    v8f acc = {};
#pragma unroll
    for (int kc = 0; kc < 8; ++kc) {
        v16h a  = frag_a(arow + kc * 32, C_);
        v16h bb = frag_b(wt + kc * 32, C_);
        acc = wmma32(a, bb, acc);
    }
    int lane = threadIdx.x & 31, col = lane & 15, hf = lane >> 4;
    int j = j0 + col;
    float bias = bqkv[j];
#pragma unroll
    for (int i = 0; i < 8; ++i) {
        int w = w0 + i + hf * 8;
        float val = acc[i] + bias;
        if (j0 < 32) {
            q[((b * H_ + h) * W_ + w) * CQ_ + j] = (_Float16)val;
        } else if (j0 < 64) {
            kk[((b * H_ + h) * W_ + w) * CQ_ + (j - 32)] = (_Float16)val;
        } else {
            int c = j - 64;
            vW[((b * H_ + h) * C_ + c) * W_ + w] = (_Float16)val;  // [b][h][c][w]
            vH[((b * W_ + w) * C_ + c) * H_ + h] = (_Float16)val;  // [b][w][c][h]
        }
    }
}

// ---------------------------------------------------------------------------
// Energies. blockIdx.y==0: row attention  E_W[b][h][w][W'] (fixed b,h)
//           blockIdx.y==1: col attention  E_H[b][w][h][H'] (fixed b,w)
// One GEMM 96x96, K=Cq=32 (single WMMA chunk). 6 waves = N tiles.
// ---------------------------------------------------------------------------
__global__ __launch_bounds__(192)
void k_energy(const _Float16* __restrict__ q, const _Float16* __restrict__ kk,
              float* __restrict__ eW, float* __restrict__ eH) {
    int line = blockIdx.x;                 // 0..383
    int mode = blockIdx.y;                 // 0 = W, 1 = H
    int b = line / 96, s = line % 96;
    const _Float16 *qb, *kb;
    int ld;
    float* e;
    if (mode == 0) {
        qb = q  + (b * H_ + s) * W_ * CQ_;       // rows = w, ld = 32
        kb = kk + (b * H_ + s) * W_ * CQ_;
        ld = CQ_;
        e = eW + line * 96 * 96;
    } else {
        qb = q  + (b * H_ * W_ + s) * CQ_;       // rows = h, ld = 96*32
        kb = kk + (b * H_ * W_ + s) * CQ_;
        ld = W_ * CQ_;
        e = eH + line * 96 * 96;
    }
    int n0 = (threadIdx.x >> 5) * 16;
    int lane = threadIdx.x & 31, col = lane & 15, hf = lane >> 4;
    for (int mt = 0; mt < 6; ++mt) {
        v16h a  = frag_a(qb + mt * 16 * ld, ld);
        v16h bb = frag_b(kb + n0 * ld, ld);      // k rows as col-major B
        v8f acc = {};
        acc = wmma32(a, bb, acc);
#pragma unroll
        for (int i = 0; i < 8; ++i)
            e[(mt * 16 + i + hf * 8) * 96 + n0 + col] = acc[i];
    }
}

// ---------------------------------------------------------------------------
// Joint softmax over [E_H(96, diag masked) ++ E_W(96)] per pixel.
// ---------------------------------------------------------------------------
__global__ void k_softmax(const float* __restrict__ eW, const float* __restrict__ eH,
                          _Float16* __restrict__ attW, _Float16* __restrict__ attH) {
    int idx = blockIdx.x * 256 + threadIdx.x;    // 36864 pixels
    int w = idx % 96;
    int t = idx / 96;
    int h = t % 96;
    int b = t / 96;
    if (b >= B_) return;
    const float* pH = eH + ((b * 96 + w) * 96 + h) * 96;
    const float* pW = eW + ((b * 96 + h) * 96 + w) * 96;
    float mx = NEG_INF;
    for (int i = 0; i < 96; ++i) { float v = (i == h) ? NEG_INF : pH[i]; mx = fmaxf(mx, v); }
    for (int i = 0; i < 96; ++i) mx = fmaxf(mx, pW[i]);
    float sum = 0.f;
    for (int i = 0; i < 96; ++i) sum += (i == h) ? 0.f : __expf(pH[i] - mx);
    for (int i = 0; i < 96; ++i) sum += __expf(pW[i] - mx);
    float inv = 1.f / sum;
    _Float16* oH = attH + ((b * 96 + w) * 96 + h) * 96;
    _Float16* oW = attW + ((b * 96 + h) * 96 + w) * 96;
    for (int i = 0; i < 96; ++i)
        oH[i] = (_Float16)((i == h) ? 0.f : __expf(pH[i] - mx) * inv);
    for (int i = 0; i < 96; ++i)
        oW[i] = (_Float16)(__expf(pW[i] - mx) * inv);
}

// ---------------------------------------------------------------------------
// out_W GEMM per (b,h): [96w x 96W'] x vW[c][W']^T -> writes NCHW f32 (raw)
// ---------------------------------------------------------------------------
__global__ __launch_bounds__(192)
void k_outW(const _Float16* __restrict__ attW, const _Float16* __restrict__ vW,
            float* __restrict__ out) {
    int line = blockIdx.x;                       // (b,h)
    int b = line / 96, h = line % 96;
    int n0 = blockIdx.y * 16;
    int m0 = (threadIdx.x >> 5) * 16;
    const _Float16* A  = attW + line * 9216;     // [w][W'] ld=96
    const _Float16* Bv = vW + (size_t)line * C_ * W_;  // [c][W'] ld=96
    v8f acc = {};
#pragma unroll
    for (int kc = 0; kc < 3; ++kc) {
        v16h a  = frag_a(A + m0 * 96 + kc * 32, 96);
        v16h bb = frag_b(Bv + n0 * 96 + kc * 32, 96);
        acc = wmma32(a, bb, acc);
    }
    int lane = threadIdx.x & 31, col = lane & 15, hf = lane >> 4;
    int c = n0 + col;
#pragma unroll
    for (int i = 0; i < 8; ++i) {
        int w = m0 + i + hf * 8;
        out[((b * C_ + c) * H_ + h) * W_ + w] = acc[i];
    }
}

// ---------------------------------------------------------------------------
// out_H GEMM per (b,w) + final fuse: out = gamma*(out_H + out_W) + x
// ---------------------------------------------------------------------------
__global__ __launch_bounds__(192)
void k_outH(const _Float16* __restrict__ attH, const _Float16* __restrict__ vH,
            const float* __restrict__ x, const float* __restrict__ gamma,
            float* __restrict__ out) {
    int line = blockIdx.x;                       // (b,w)
    int b = line / 96, w = line % 96;
    int n0 = blockIdx.y * 16;
    int m0 = (threadIdx.x >> 5) * 16;
    const _Float16* A  = attH + line * 9216;     // [h][H'] ld=96
    const _Float16* Bv = vH + (size_t)line * C_ * H_;  // [c][H'] ld=96
    v8f acc = {};
#pragma unroll
    for (int kc = 0; kc < 3; ++kc) {
        v16h a  = frag_a(A + m0 * 96 + kc * 32, 96);
        v16h bb = frag_b(Bv + n0 * 96 + kc * 32, 96);
        acc = wmma32(a, bb, acc);
    }
    float g = gamma[0];
    int lane = threadIdx.x & 31, col = lane & 15, hf = lane >> 4;
    int c = n0 + col;
#pragma unroll
    for (int i = 0; i < 8; ++i) {
        int h = m0 + i + hf * 8;
        int o = ((b * C_ + c) * H_ + h) * W_ + w;
        out[o] = g * (acc[i] + out[o]) + x[o];
    }
}

// ---------------------------------------------------------------------------
extern "C" void kernel_launch(void* const* d_in, const int* in_sizes, int n_in,
                              void* d_out, int out_size, void* d_ws, size_t ws_size,
                              hipStream_t stream) {
    const float* x    = (const float*)d_in[0];
    const float* wms  = (const float*)d_in[1];
    const float* bms0 = (const float*)d_in[2];
    const float* wq   = (const float*)d_in[3];
    const float* bq   = (const float*)d_in[4];
    const float* wk   = (const float*)d_in[5];
    const float* bk   = (const float*)d_in[6];
    const float* wv   = (const float*)d_in[7];
    const float* bv   = (const float*)d_in[8];
    const float* gm   = (const float*)d_in[9];
    float* out = (float*)d_out;

    char* p = (char*)d_ws;
    auto alloc = [&](size_t bytes) -> char* {
        char* r = p;
        p += (bytes + 255) & ~(size_t)255;
        return r;
    };
    _Float16* xp   = (_Float16*)alloc((size_t)B_ * HP_ * HP_ * C_ * 2);  // 21.3 MB
    _Float16* wpk  = (_Float16*)alloc((size_t)NT_ * C_ * C_ * 2);        //  3.5 MB
    _Float16* wqkv = (_Float16*)alloc((size_t)320 * C_ * 2);
    float*    bqkv = (float*)   alloc(320 * 4);
    float*    bms  = (float*)   alloc(C_ * 4);
    _Float16* ms   = (_Float16*)alloc((size_t)B_ * H_ * W_ * C_ * 2);    // 18.9 MB
    _Float16* q    = (_Float16*)alloc((size_t)B_ * H_ * W_ * CQ_ * 2);
    _Float16* kk   = (_Float16*)alloc((size_t)B_ * H_ * W_ * CQ_ * 2);
    _Float16* vW   = (_Float16*)alloc((size_t)B_ * H_ * C_ * W_ * 2);    // 18.9 MB
    _Float16* vH   = (_Float16*)alloc((size_t)B_ * W_ * C_ * H_ * 2);    // 18.9 MB
    float*    eW   = (float*)   alloc((size_t)B_ * H_ * W_ * 96 * 4);    // 14.2 MB
    float*    eH   = (float*)   alloc((size_t)B_ * H_ * W_ * 96 * 4);    // 14.2 MB
    _Float16* attW = (_Float16*)alloc((size_t)B_ * H_ * W_ * 96 * 2);    //  7.1 MB
    _Float16* attH = (_Float16*)alloc((size_t)B_ * H_ * W_ * 96 * 2);    //  7.1 MB

    // prep / repack
    k_pad<<<(B_ * HP_ * HP_ * C_ + 255) / 256, 256, 0, stream>>>(x, xp);
    k_packw<<<(NT_ * C_ * C_ + 255) / 256, 256, 0, stream>>>(wms, wpk);
    k_packqkv<<<(320 * 256 + 255) / 256, 256, 0, stream>>>(wq, bq, wk, bk, wv, bv,
                                                           bms0, wqkv, bqkv, bms);
    // multi-scale dilated conv (implicit GEMM, 2x2 blocked, async-staged)
    k_conv<<<B_ * H_ * (W_ / 32), 256, 0, stream>>>(xp, wpk, bms, ms);
    // fused q/k/v projections
    k_qkv<<<dim3(B_ * H_ * (W_ / 16), 20), 32, 0, stream>>>(ms, wqkv, bqkv, q, kk, vW, vH);
    // criss-cross energies (y=0: row/W, y=1: col/H)
    k_energy<<<dim3(B_ * 96, 2), 192, 0, stream>>>(q, kk, eW, eH);
    // joint softmax with diagonal mask
    k_softmax<<<(B_ * H_ * W_ + 255) / 256, 256, 0, stream>>>(eW, eH, attW, attH);
    // attention outputs + residual fuse
    k_outW<<<dim3(B_ * H_, 16), 192, 0, stream>>>(attW, vW, out);
    k_outH<<<dim3(B_ * W_, 16), 192, 0, stream>>>(attH, vH, x, gm, out);
}
